// Down_Block_6614249636481
// MI455X (gfx1250) — matrix operand order repaired and checked by reading, not compiled, not run on hardware
//
#include <hip/hip_runtime.h>
#include <hip/hip_bf16.h>

// ---------------------------------------------------------------------------
// Problem constants (from the reference)
// ---------------------------------------------------------------------------
#define B_    2
#define C_    96
#define Dd_   8
#define Hh_   48
#define Ww_   48
#define L_    18432            // Dd_*Hh_*Ww_
#define M_    36864            // B_*L_
#define DIN   192              // D_INNER
#define DST   16               // D_STATE
#define DTR   6                // DT_RANK
#define XP_N  48               // x_proj output padded 38 -> 48 (multiple of 16)

typedef __attribute__((ext_vector_type(16))) _Float16 v16h;
typedef __attribute__((ext_vector_type(8)))  _Float16 v8h;
typedef __attribute__((ext_vector_type(8)))  float    v8f;

// ---------------------------------------------------------------------------
// WMMA GEMM:  C[M,N] = A[M,K] (f16, row-major) * Bw[N,K]^T (f16, "weights" layout)
// Each wave computes a 64x16 tile (4 m-tiles sharing one B fragment).
// EPI: 0 = none, 1 = +bias, 2 = +bias then exact GELU.  OUTH: write f16 else f32.
// ---------------------------------------------------------------------------
template <int EPI, bool OUTH>
__global__ __launch_bounds__(256) void gemm_wmma_kernel(
    const _Float16* __restrict__ A, const _Float16* __restrict__ Bw,
    const float* __restrict__ bias, float* __restrict__ Cf,
    _Float16* __restrict__ Ch, int M, int N, int K)
{
  constexpr int MREP = 4;
  const int lane = threadIdx.x & 31;
  const int wave = threadIdx.x >> 5;
  const int mbase = (blockIdx.x * 8 + wave) * (16 * MREP);
  const int n0 = blockIdx.y * 16;
  if (mbase >= M) return;                       // wave-uniform: EXEC stays all-1s

  const int row  = lane & 15;                   // M-row (A) / N-row (B) / N-col (C)
  const int half = lane >> 4;

  const _Float16* __restrict__ bp  = Bw + (size_t)(n0 + row) * K + half * 16;
  const _Float16* __restrict__ ap0 = A  + (size_t)(mbase + row) * K + half * 16;

  v8f acc[MREP] = {};
  for (int k0 = 0; k0 < K; k0 += 32) {
    // B fragment (32x16, 16-bit): lanes 0-15 -> K=k0..k0+15, lanes 16-31 -> +16
    v16h bf = *(const v16h*)(bp + k0);
#pragma unroll
    for (int r = 0; r < MREP; ++r) {
      const _Float16* ap = ap0 + (size_t)r * 16 * K;
      // A fragment (16x32, 16-bit): lanes 0-15 -> K=k0..k0+15, lanes 16-31 -> +16
      v16h af = *(const v16h*)(ap + k0);
      acc[r] = __builtin_amdgcn_wmma_f32_16x16x32_f16(
          /*neg_a=*/false, af, /*neg_b=*/false, bf,
          /*c_mod=*/(short)0, acc[r], /*reuse_a=*/false, /*reuse_b=*/false);
    }
  }

  const int n = n0 + row;
  const float bv = (EPI >= 1) ? bias[n] : 0.0f;
#pragma unroll
  for (int r = 0; r < MREP; ++r) {
    const int mt = mbase + r * 16;
#pragma unroll
    for (int j = 0; j < 8; ++j) {
      const int m = mt + half * 8 + j;          // C/D: VGPR j, half -> M row
      float v = acc[r][j] + bv;
      if (EPI == 2) v = 0.5f * v * (1.0f + erff(v * 0.70710678118654752f));
      if (OUTH) Ch[(size_t)m * N + n] = (_Float16)v;
      else      Cf[(size_t)m * N + n] = v;
    }
  }
}

// ---------------------------------------------------------------------------
// Weight prep
// ---------------------------------------------------------------------------
__global__ void cvt_f32_f16_kernel(const float* __restrict__ s,
                                   _Float16* __restrict__ d, int n)
{
  int i = blockIdx.x * blockDim.x + threadIdx.x;
  if (i < n) d[i] = (_Float16)s[i];
}

__global__ void xproj_pad_kernel(const float* __restrict__ s,   // [38,192]
                                 _Float16* __restrict__ d)      // [48,192]
{
  int i = blockIdx.x * blockDim.x + threadIdx.x;
  if (i >= XP_N * DIN) return;
  int r = i / DIN, c = i % DIN;
  d[i] = (_Float16)((r < DTR + 2 * DST) ? s[r * DIN + c] : 0.0f);
}

// ---------------------------------------------------------------------------
// Depthwise 3D conv (3x7x7, pad 1,3,3) + bias -> hconv [B,C,L]
// ---------------------------------------------------------------------------
__global__ void dwconv3d_kernel(const float* __restrict__ x,
                                const float* __restrict__ w,
                                const float* __restrict__ bias,
                                float* __restrict__ outp)
{
  int idx = blockIdx.x * blockDim.x + threadIdx.x;
  if (idx >= B_ * C_ * L_) return;
  int ww = idx % Ww_;  int t = idx / Ww_;
  int hh = t % Hh_;    t /= Hh_;
  int dd = t % Dd_;    t /= Dd_;
  int c  = t % C_;     int b = t / C_;
  const float* xb = x + (size_t)(b * C_ + c) * L_;
  const float* wc = w + c * 147;
  float acc = bias[c];
  for (int kd = 0; kd < 3; ++kd) {
    int zd = dd + kd - 1; if (zd < 0 || zd >= Dd_) continue;
    for (int kh = 0; kh < 7; ++kh) {
      int zh = hh + kh - 3; if (zh < 0 || zh >= Hh_) continue;
      for (int kw = 0; kw < 7; ++kw) {
        int zw = ww + kw - 3; if (zw < 0 || zw >= Ww_) continue;
        acc += xb[(zd * Hh_ + zh) * Ww_ + zw] * wc[(kd * 7 + kh) * 7 + kw];
      }
    }
  }
  outp[idx] = acc;
}

// ---------------------------------------------------------------------------
// GroupNorm (8 groups of 12 channels): stats then apply+transpose to f16 [M,C]
// ---------------------------------------------------------------------------
__global__ void gn_stats_kernel(const float* __restrict__ h, float* __restrict__ stats)
{
  const int bg = blockIdx.x;                 // 0..15 : b*8+g
  const int b = bg >> 3, g = bg & 7;
  const float* base = h + (size_t)(b * C_ + g * 12) * L_;
  float s = 0.f, sq = 0.f;
  for (int i = threadIdx.x; i < 12 * L_; i += blockDim.x) {
    float v = base[i]; s += v; sq += v * v;
  }
  __shared__ float ss[256], sg[256];
  ss[threadIdx.x] = s; sg[threadIdx.x] = sq; __syncthreads();
  for (int st = 128; st; st >>= 1) {
    if (threadIdx.x < st) { ss[threadIdx.x] += ss[threadIdx.x + st];
                            sg[threadIdx.x] += sg[threadIdx.x + st]; }
    __syncthreads();
  }
  if (threadIdx.x == 0) {
    float inv = 1.0f / (12.0f * L_);
    float mu = ss[0] * inv;
    float var = sg[0] * inv - mu * mu;
    stats[bg * 2 + 0] = mu;
    stats[bg * 2 + 1] = rsqrtf(var + 1e-5f);
  }
}

__global__ void gn_apply_kernel(const float* __restrict__ h,
                                const float* __restrict__ stats,
                                const float* __restrict__ g,
                                const float* __restrict__ bta,
                                _Float16* __restrict__ outp)   // [M, C]
{
  int idx = blockIdx.x * blockDim.x + threadIdx.x;
  if (idx >= M_ * C_) return;
  int m = idx / C_, c = idx % C_;
  int b = m / L_, l = m % L_;
  int grp = c / 12;
  float mu   = stats[(b * 8 + grp) * 2 + 0];
  float rstd = stats[(b * 8 + grp) * 2 + 1];
  float v = h[(size_t)(b * C_ + c) * L_ + l];
  outp[idx] = (_Float16)((v - mu) * rstd * g[c] + bta[c]);
}

// ---------------------------------------------------------------------------
// LayerNorm over C=96: one wave per row, 3 channels per lane, shfl_xor reduce.
// MODE 0: input = x (transposed read).  MODE 1: input = outm + skip*x.
// ---------------------------------------------------------------------------
template <int MODE>
__global__ __launch_bounds__(256) void ln_kernel(
    const float* __restrict__ xin,       // [B,C,L]
    const float* __restrict__ outm,      // [M,C] (MODE 1)
    const float* __restrict__ skip,      // [1]   (MODE 1)
    const float* __restrict__ g, const float* __restrict__ bta,
    _Float16* __restrict__ outh)         // [M,C]
{
  const int wave = threadIdx.x >> 5, lane = threadIdx.x & 31;
  const int m = blockIdx.x * 8 + wave;
  const int b = m / L_, l = m % L_;
  const float* base = xin + (size_t)b * C_ * L_ + l;
  float v0 = base[(size_t)(lane +  0) * L_];
  float v1 = base[(size_t)(lane + 32) * L_];
  float v2 = base[(size_t)(lane + 64) * L_];
  if (MODE == 1) {
    const float sk = skip[0];
    const float* om = outm + (size_t)m * C_;
    v0 = om[lane +  0] + sk * v0;
    v1 = om[lane + 32] + sk * v1;
    v2 = om[lane + 64] + sk * v2;
  }
  float s = v0 + v1 + v2, sq = v0 * v0 + v1 * v1 + v2 * v2;
  for (int off = 16; off; off >>= 1) {
    s  += __shfl_xor(s,  off, 32);
    sq += __shfl_xor(sq, off, 32);
  }
  float mu   = s * (1.0f / 96.0f);
  float var  = sq * (1.0f / 96.0f) - mu * mu;
  float rstd = rsqrtf(var + 1e-5f);
  _Float16* o = outh + (size_t)m * C_;
  o[lane +  0] = (_Float16)((v0 - mu) * rstd * g[lane +  0] + bta[lane +  0]);
  o[lane + 32] = (_Float16)((v1 - mu) * rstd * g[lane + 32] + bta[lane + 32]);
  o[lane + 64] = (_Float16)((v2 - mu) * rstd * g[lane + 64] + bta[lane + 64]);
}

// ---------------------------------------------------------------------------
// Causal depthwise conv1d (K=4) + SiLU -> u (f32 + f16), [M, DIN]
// xm lives in xz[:, 0:192]
// ---------------------------------------------------------------------------
__global__ void conv1d_silu_kernel(const float* __restrict__ xz,
                                   const float* __restrict__ w,
                                   const float* __restrict__ bias,
                                   float* __restrict__ uf,
                                   _Float16* __restrict__ uh)
{
  int idx = blockIdx.x * blockDim.x + threadIdx.x;
  if (idx >= M_ * DIN) return;
  int m = idx / DIN, d = idx % DIN;
  int b = m / L_, l = m % L_;
  float acc = bias[d];
#pragma unroll
  for (int k = 0; k < 4; ++k) {
    int ll = l - 3 + k;
    if (ll >= 0) acc += w[d * 4 + k] * xz[(size_t)(b * L_ + ll) * (2 * DIN) + d];
  }
  float u = acc / (1.0f + __expf(-acc));
  uf[idx] = u;
  uh[idx] = (_Float16)u;
}

// dt = softplus(x_dbl[:, 0:6] @ dt_proj_w.T + dt_proj_b)   -> [M, DIN]
__global__ void dt_kernel(const float* __restrict__ xdbl,
                          const float* __restrict__ w,     // [DIN, 6]
                          const float* __restrict__ bias,
                          float* __restrict__ dt)
{
  int idx = blockIdx.x * blockDim.x + threadIdx.x;
  if (idx >= M_ * DIN) return;
  int m = idx / DIN, d = idx % DIN;
  float acc = bias[d];
#pragma unroll
  for (int r = 0; r < DTR; ++r) acc += w[d * DTR + r] * xdbl[(size_t)m * XP_N + r];
  dt[idx] = (acc > 20.0f) ? acc : log1pf(__expf(acc));
}

// ---------------------------------------------------------------------------
// Selective scan: lane = (d_local*16 + s), one wave handles 2 channels x 16
// states; 192 waves total (2 batches x 96).  Serial over L with shfl reduce.
// ---------------------------------------------------------------------------
__global__ __launch_bounds__(32) void scan_kernel(
    const float* __restrict__ dt,    // [M, DIN]
    const float* __restrict__ uf,    // [M, DIN]
    const float* __restrict__ xdbl,  // [M, 48]: 6..21 = B, 22..37 = C
    const float* __restrict__ A_log, // [DIN, 16]
    float* __restrict__ ybuf)        // [M, DIN]
{
  const int wid = blockIdx.x;            // 0..191
  const int b = wid / (DIN / 2), dp = wid % (DIN / 2);
  const int lane = threadIdx.x;
  const int s = lane & 15;
  const int d = dp * 2 + (lane >> 4);
  const float Av = -__expf(A_log[d * DST + s]);
  float h = 0.0f;
  const size_t m0 = (size_t)b * L_;
  for (int l = 0; l < L_; ++l) {
    const size_t m = m0 + l;
    float dtv = dt[m * DIN + d];
    float uv  = uf[m * DIN + d];
    float Bv  = xdbl[m * XP_N + DTR + s];
    float Cv  = xdbl[m * XP_N + DTR + DST + s];
    h = __expf(dtv * Av) * h + (dtv * uv) * Bv;
    float yc = h * Cv;
    yc += __shfl_xor(yc, 1, 16);
    yc += __shfl_xor(yc, 2, 16);
    yc += __shfl_xor(yc, 4, 16);
    yc += __shfl_xor(yc, 8, 16);
    if (s == 0) ybuf[m * DIN + d] = yc;
  }
}

// y = (ys + u*Dp) * silu(z) -> f16 [M, DIN];   z = xz[:, 192:384]
__global__ void gate_kernel(const float* __restrict__ ybuf,
                            const float* __restrict__ uf,
                            const float* __restrict__ Dp,
                            const float* __restrict__ xz,
                            _Float16* __restrict__ yh)
{
  int idx = blockIdx.x * blockDim.x + threadIdx.x;
  if (idx >= M_ * DIN) return;
  int m = idx / DIN, d = idx % DIN;
  float y = ybuf[idx] + uf[idx] * Dp[d];
  float z = xz[(size_t)m * (2 * DIN) + DIN + d];
  y *= z / (1.0f + __expf(-z));
  yh[idx] = (_Float16)y;
}

// d_out[b,c,l] = conv_out[m,c] + mamba[m,c]
__global__ void final_add_kernel(const float* __restrict__ convout,
                                 const float* __restrict__ mamba,
                                 float* __restrict__ outp)
{
  int idx = blockIdx.x * blockDim.x + threadIdx.x;
  if (idx >= B_ * C_ * L_) return;
  int l = idx % L_; int t = idx / L_;
  int c = t % C_;   int b = t / C_;
  size_t m = (size_t)b * L_ + l;
  outp[idx] = convout[m * C_ + c] + mamba[m * C_ + c];
}

// ---------------------------------------------------------------------------
// Launcher
// ---------------------------------------------------------------------------
extern "C" void kernel_launch(void* const* d_in, const int* in_sizes, int n_in,
                              void* d_out, int out_size, void* d_ws, size_t ws_size,
                              hipStream_t stream)
{
  (void)in_sizes; (void)n_in; (void)out_size; (void)ws_size;
  const float* x      = (const float*)d_in[0];
  const float* dw_w   = (const float*)d_in[1];
  const float* dw_b   = (const float*)d_in[2];
  const float* gn_g   = (const float*)d_in[3];
  const float* gn_b   = (const float*)d_in[4];
  const float* pw1_w  = (const float*)d_in[5];
  const float* pw1_b  = (const float*)d_in[6];
  const float* pw2_w  = (const float*)d_in[7];
  const float* pw2_b  = (const float*)d_in[8];
  const float* ln_g   = (const float*)d_in[9];
  const float* ln_b   = (const float*)d_in[10];
  const float* skip   = (const float*)d_in[11];
  const float* inpj_w = (const float*)d_in[12];
  const float* c1_w   = (const float*)d_in[13];
  const float* c1_b   = (const float*)d_in[14];
  const float* xpj_w  = (const float*)d_in[15];
  const float* dtp_w  = (const float*)d_in[16];
  const float* dtp_b  = (const float*)d_in[17];
  const float* A_log  = (const float*)d_in[18];
  const float* Dp     = (const float*)d_in[19];
  const float* opj_w  = (const float*)d_in[20];
  const float* pj_w   = (const float*)d_in[21];
  const float* pj_b   = (const float*)d_in[22];
  float* outp = (float*)d_out;

  char* ws = (char*)d_ws;
  size_t off = 0;
  auto alloc = [&](size_t bytes) {
    size_t r = off; off = (off + bytes + 255) & ~(size_t)255; return r;
  };

  // fp16 weight pool
  _Float16* pw1h  = (_Float16*)(ws + alloc(sizeof(_Float16) * 384 * 96));
  _Float16* pw2h  = (_Float16*)(ws + alloc(sizeof(_Float16) * 96 * 384));
  _Float16* inph  = (_Float16*)(ws + alloc(sizeof(_Float16) * 384 * 96));
  _Float16* xpjh  = (_Float16*)(ws + alloc(sizeof(_Float16) * XP_N * DIN));
  _Float16* opjh  = (_Float16*)(ws + alloc(sizeof(_Float16) * 96 * DIN));
  _Float16* pjh   = (_Float16*)(ws + alloc(sizeof(_Float16) * 96 * 96));
  // activations (with lifetime-based reuse)
  float*    hconv = (float*)(ws + alloc(sizeof(float) * (size_t)M_ * C_));
  float*    gstat = (float*)(ws + alloc(sizeof(float) * 32));
  _Float16* hn    = (_Float16*)(ws + alloc(sizeof(_Float16) * (size_t)M_ * C_));
  _Float16* act1  = (_Float16*)(ws + alloc(sizeof(_Float16) * (size_t)M_ * 384));
  float*    cvout = (float*)(ws + alloc(sizeof(float) * (size_t)M_ * C_));
  _Float16* xn    = (_Float16*)(ws + alloc(sizeof(_Float16) * (size_t)M_ * C_));
  float*    xz    = (float*)(ws + alloc(sizeof(float) * (size_t)M_ * 384));
  float*    uf    = (float*)(ws + alloc(sizeof(float) * (size_t)M_ * DIN));
  _Float16* uh    = (_Float16*)(ws + alloc(sizeof(_Float16) * (size_t)M_ * DIN));
  float*    xdbl  = (float*)(ws + alloc(sizeof(float) * (size_t)M_ * XP_N));
  float*    dtb   = (float*)(ws + alloc(sizeof(float) * (size_t)M_ * DIN));
  // reuse (same byte footprint, disjoint lifetimes)
  float*    ybuf  = (float*)act1;      // act1 dead after pw2 GEMM
  _Float16* yh    = (_Float16*)uh;     // uh dead after x_proj GEMM
  float*    outm  = (float*)hconv;     // hconv dead after gn_apply
  _Float16* xm2n  = (_Float16*)xn;     // xn dead after in_proj GEMM
  float*    mamba = (float*)dtb;       // dt dead after scan

  const int T = 256;
  auto blk = [](int n, int t) { return (n + t - 1) / t; };

  // -- weight prep -----------------------------------------------------------
  cvt_f32_f16_kernel<<<blk(384 * 96, T), T, 0, stream>>>(pw1_w,  pw1h, 384 * 96);
  cvt_f32_f16_kernel<<<blk(96 * 384, T), T, 0, stream>>>(pw2_w,  pw2h, 96 * 384);
  cvt_f32_f16_kernel<<<blk(384 * 96, T), T, 0, stream>>>(inpj_w, inph, 384 * 96);
  cvt_f32_f16_kernel<<<blk(96 * DIN, T), T, 0, stream>>>(opj_w,  opjh, 96 * DIN);
  cvt_f32_f16_kernel<<<blk(96 * 96, T), T, 0, stream>>>(pj_w,   pjh,  96 * 96);
  xproj_pad_kernel<<<blk(XP_N * DIN, T), T, 0, stream>>>(xpj_w, xpjh);

  // -- conv branch -----------------------------------------------------------
  dwconv3d_kernel<<<blk(B_ * C_ * L_, T), T, 0, stream>>>(x, dw_w, dw_b, hconv);
  gn_stats_kernel<<<16, 256, 0, stream>>>(hconv, gstat);
  gn_apply_kernel<<<blk(M_ * C_, T), T, 0, stream>>>(hconv, gstat, gn_g, gn_b, hn);
  gemm_wmma_kernel<2, true ><<<dim3(M_ / 512, 384 / 16), 256, 0, stream>>>(
      hn, pw1h, pw1_b, nullptr, act1, M_, 384, 96);                 // pw1+GELU
  gemm_wmma_kernel<1, false><<<dim3(M_ / 512, 96 / 16), 256, 0, stream>>>(
      act1, pw2h, pw2_b, cvout, nullptr, M_, 96, 384);              // pw2+bias

  // -- mamba branch ----------------------------------------------------------
  ln_kernel<0><<<M_ / 8, 256, 0, stream>>>(x, nullptr, nullptr, ln_g, ln_b, xn);
  gemm_wmma_kernel<0, false><<<dim3(M_ / 512, 384 / 16), 256, 0, stream>>>(
      xn, inph, nullptr, xz, nullptr, M_, 384, 96);                 // in_proj
  conv1d_silu_kernel<<<blk(M_ * DIN, T), T, 0, stream>>>(xz, c1_w, c1_b, uf, uh);
  gemm_wmma_kernel<0, false><<<dim3(M_ / 512, XP_N / 16), 256, 0, stream>>>(
      uh, xpjh, nullptr, xdbl, nullptr, M_, XP_N, DIN);             // x_proj
  dt_kernel<<<blk(M_ * DIN, T), T, 0, stream>>>(xdbl, dtp_w, dtp_b, dtb);
  scan_kernel<<<B_ * (DIN / 2), 32, 0, stream>>>(dtb, uf, xdbl, A_log, ybuf);
  gate_kernel<<<blk(M_ * DIN, T), T, 0, stream>>>(ybuf, uf, Dp, xz, yh);
  gemm_wmma_kernel<0, false><<<dim3(M_ / 512, 96 / 16), 256, 0, stream>>>(
      yh, opjh, nullptr, outm, nullptr, M_, 96, DIN);               // out_proj
  ln_kernel<1><<<M_ / 8, 256, 0, stream>>>(x, outm, skip, ln_g, ln_b, xm2n);
  gemm_wmma_kernel<1, false><<<dim3(M_ / 512, 96 / 16), 256, 0, stream>>>(
      xm2n, pjh, pj_b, mamba, nullptr, M_, 96, 96);                 // proj+bias

  // -- combine ---------------------------------------------------------------
  final_add_kernel<<<blk(B_ * C_ * L_, T), T, 0, stream>>>(cvout, mamba, outp);
}
